// NemotronHMamba2Mixer_57647051047632
// MI455X (gfx1250) — matrix operand order repaired and checked by reading, not compiled, not run on hardware
//
#include <hip/hip_runtime.h>
#include <hip/hip_bf16.h>

// ---------------- problem constants (from reference) ----------------
constexpr int L_    = 2048;
constexpr int HIDN  = 4096;
constexpr int H_    = 128;
constexpr int P_    = 64;
constexpr int N_    = 128;
constexpr int G_    = 8;
constexpr int K_    = 4;
constexpr int CS_   = 256;
constexpr int INTER = H_ * P_;             // 8192
constexpr int CONVC = INTER + 2 * G_ * N_; // 10240
constexpr int PROJ  = INTER + CONVC + H_;  // 18560
constexpr int GS_   = INTER / G_;          // 1024
constexpr int NC_   = L_ / CS_;            // 8
constexpr float EPSF = 1e-5f;

// ---------------- CDNA5 vector types ----------------
typedef __attribute__((ext_vector_type(16))) __bf16 v16bf;
typedef __attribute__((ext_vector_type(8)))  __bf16 v8bf;
typedef __attribute__((ext_vector_type(8)))  float  v8f;

static __device__ __forceinline__ __bf16 f2bf(float f) {
    union { float f; unsigned u; } v; v.f = f;
    unsigned r = v.u + 0x7FFFu + ((v.u >> 16) & 1u);   // round-to-nearest-even
    union { unsigned short s; __bf16 b; } o;
    o.s = (unsigned short)(r >> 16);
    return o.b;
}

// async global->LDS copy of one 16B chunk per lane (512B per wave-instruction),
// tracked by ASYNCcnt (CDNA5 async-tensor path).
static __device__ __forceinline__ void async_b128(unsigned lds_off, const void* gaddr) {
    asm volatile("global_load_async_to_lds_b128 %0, %1, off"
                 :: "v"(lds_off), "v"(gaddr) : "memory");
}
static __device__ __forceinline__ void wait_async0() {
    asm volatile("s_wait_asynccnt 0x0" ::: "memory");
}
static __device__ __forceinline__ unsigned lds_off_of(const void* p) {
    return (unsigned)(unsigned long long)p;    // generic LDS addr: low 32 bits = LDS offset
}

// ---------------- batched bf16 WMMA GEMM (C = A * B^T), LDS double-buffered ----
// A: [M, K] rows lda apart (bf16, K contiguous)
// B: [N, K] rows ldb apart (bf16, K contiguous)
// C: [M, N] rows ldc apart (f32)
// Block = 256 threads = 8 waves as 4(M) x 2(N); block tile 64x128, BK = 128.
// Each wave owns a 16x64 output tile (4 accumulators sharing one A fragment).
// B fragments are register double-buffered across k-chunks so ds_loads stay
// ahead of the WMMAs (partial s_wait_dscnt instead of 0).
#define BM 64
#define BN 128
#define BK 128
#define KC (BK / 32)

__global__ __launch_bounds__(256)
void gemm_bf16_nt(const __bf16* __restrict__ A,
                  const __bf16* __restrict__ B,
                  float* __restrict__ C,
                  int M, int N, int Kd,
                  int lda, int ldb, int ldc,
                  long long strA, long long strB, long long strC,
                  int accum) {
    __shared__ __align__(16) __bf16 As[2][BM][BK];   // 32 KB
    __shared__ __align__(16) __bf16 Bs[2][BN][BK];   // 64 KB

    const long long z = blockIdx.z;
    A += z * strA;  B += z * strB;  C += z * strC;

    const int tid   = threadIdx.x;
    // scalar wave id: keeps the `active` guard an SALU branch (EXEC stays all-1s)
    const int waveS = __builtin_amdgcn_readfirstlane(tid) >> 5;
    const int lane  = tid & 31;
    const int m0 = blockIdx.x * BM;
    const int n0 = blockIdx.y * BN;

    const int wm = (waveS >> 1) * 16;     // wave's M offset in block tile
    const int wn = (waveS & 1) * 64;      // wave's N offset in block tile
    const bool active = (n0 + wn + 64 <= N);  // scalar; M always divides exactly

    // stage one BK-slice of A and B into LDS buffer `buf` via async DMA
    auto stage = [&](int buf, int k0) {
        // A tile: BM*BK bf16 = 16KB = 1024 x 16B chunks, 4 per thread
#pragma unroll
        for (int it = 0; it < 4; ++it) {
            int idx = it * 256 + tid;
            int row = idx >> 4;            // BK/8 = 16 chunks per row
            int col = (idx & 15) * 8;
            int gr  = m0 + row; if (gr >= M) gr = M - 1;   // clamp (keep EXEC full)
            async_b128(lds_off_of(&As[buf][row][col]),
                       A + (size_t)gr * lda + k0 + col);
        }
        // B tile: BN*BK bf16 = 32KB = 2048 x 16B chunks, 8 per thread
#pragma unroll
        for (int it = 0; it < 8; ++it) {
            int idx = it * 256 + tid;
            int row = idx >> 4;
            int col = (idx & 15) * 8;
            int gr  = n0 + row; if (gr >= N) gr = N - 1;
            async_b128(lds_off_of(&Bs[buf][row][col]),
                       B + (size_t)gr * ldb + k0 + col);
        }
    };

    const int arow = wm + (lane & 15);
    const int brow = wn + (lane & 15);
    const int kAo  = (lane < 16) ? 0 : 16;   // A frag: 16 contiguous K (matches B)
    const int kBo  = (lane < 16) ? 0 : 16;   // B frag: 16 contiguous K (matches A)

    v8f acc[4] = {};
    const int KT = Kd / BK;

    stage(0, 0);
    wait_async0();
    __syncthreads();

    for (int kt = 0; kt < KT; ++kt) {
        const int cur = kt & 1;
        if (kt + 1 < KT) stage(cur ^ 1, (kt + 1) * BK);   // DMA overlaps compute
        if (active) {
            // hoist all A fragments of this stage
            v16bf afrag[KC];
#pragma unroll
            for (int kc = 0; kc < KC; ++kc) {
                const int ko = kc * 32;
                v8bf alo = *(const v8bf*)&As[cur][arow][ko + kAo];
                v8bf ahi = *(const v8bf*)&As[cur][arow][ko + 8 + kAo];
                afrag[kc] = __builtin_shufflevector(alo, ahi,
                             0,1,2,3,4,5,6,7,8,9,10,11,12,13,14,15);
            }
            // register double-buffered B fragments: load kc+1 before wmma(kc)
            v16bf bb[2][4];
#pragma unroll
            for (int t = 0; t < 4; ++t)
                bb[0][t] = *(const v16bf*)&Bs[cur][brow + t * 16][kBo];
#pragma unroll
            for (int kc = 0; kc < KC; ++kc) {
                const int pb = kc & 1;
                if (kc + 1 < KC) {
                    const int ko = (kc + 1) * 32;
#pragma unroll
                    for (int t = 0; t < 4; ++t)
                        bb[pb ^ 1][t] = *(const v16bf*)&Bs[cur][brow + t * 16][ko + kBo];
                }
                acc[0] = __builtin_amdgcn_wmma_f32_16x16x32_bf16(
                             false, afrag[kc], false, bb[pb][0], (short)0, acc[0], false, false);
                acc[1] = __builtin_amdgcn_wmma_f32_16x16x32_bf16(
                             false, afrag[kc], false, bb[pb][1], (short)0, acc[1], false, false);
                acc[2] = __builtin_amdgcn_wmma_f32_16x16x32_bf16(
                             false, afrag[kc], false, bb[pb][2], (short)0, acc[2], false, false);
                acc[3] = __builtin_amdgcn_wmma_f32_16x16x32_bf16(
                             false, afrag[kc], false, bb[pb][3], (short)0, acc[3], false, false);
            }
        }
        if (kt + 1 < KT) wait_async0();
        __syncthreads();
    }

    if (active) {
        // C/D layout: VGPR r holds (M = r + 8*(lane/16), N = lane%16)
        const int nn = lane & 15;
        const int mo = (lane >> 4) * 8;
#pragma unroll
        for (int t = 0; t < 4; ++t)
#pragma unroll
            for (int r = 0; r < 8; ++r) {
                size_t off = (size_t)(m0 + wm + mo + r) * ldc
                           + (size_t)(n0 + wn + t * 16 + nn);
                if (accum) C[off] += acc[t][r]; else C[off] = acc[t][r];
            }
    }
}

// ---------------- elementwise / support kernels ----------------

__global__ void cvt_f32_bf16(const float* __restrict__ src, __bf16* __restrict__ dst, long long n) {
    for (long long i = (long long)blockIdx.x * blockDim.x + threadIdx.x; i < n;
         i += (long long)gridDim.x * blockDim.x)
        dst[i] = f2bf(src[i]);
}

// depthwise causal conv (K=4) + bias + silu over the xBC slice of proj
__global__ void conv_silu_kernel(const float* __restrict__ proj,
                                 const float* __restrict__ conv_w,
                                 const float* __restrict__ conv_b,
                                 float* __restrict__ xBCc) {
    long long idx = (long long)blockIdx.x * blockDim.x + threadIdx.x;
    if (idx >= (long long)L_ * CONVC) return;
    int c = (int)(idx % CONVC);
    int l = (int)(idx / CONVC);
    float acc = conv_b[c];
#pragma unroll
    for (int k = 0; k < K_; ++k) {
        int ls = l - (K_ - 1) + k;
        if (ls >= 0) acc += proj[(size_t)ls * PROJ + INTER + c] * conv_w[c * K_ + k];
    }
    xBCc[(size_t)l * CONVC + c] = acc / (1.f + expf(-acc));   // silu
}

// per (chunk, head): softplus(dt+bias), dA = dt*A, chunk-local inclusive cumsum
__global__ void dt_cum_kernel(const float* __restrict__ proj,
                              const float* __restrict__ dt_bias,
                              const float* __restrict__ A_log,
                              float* __restrict__ dt_sp,
                              float* __restrict__ cum,
                              float* __restrict__ cdec) {
    int c = blockIdx.x / H_;
    int h = blockIdx.x % H_;
    int i = threadIdx.x;                      // CS_ threads
    int l = c * CS_ + i;
    float dtv = proj[(size_t)l * PROJ + INTER + CONVC + h] + dt_bias[h];
    float sp  = (dtv > 20.f) ? dtv : log1pf(expf(dtv));
    sp = fmaxf(sp, 0.f);
    float dA = sp * (-expf(A_log[h]));
    __shared__ float sbuf[CS_];
    sbuf[i] = dA;
    __syncthreads();
    for (int off = 1; off < CS_; off <<= 1) { // Hillis-Steele inclusive scan
        float v = (i >= off) ? sbuf[i - off] : 0.f;
        __syncthreads();
        sbuf[i] += v;
        __syncthreads();
    }
    dt_sp[(size_t)l * H_ + h] = sp;
    cum[(size_t)l * H_ + h]   = sbuf[i];
    if (i == CS_ - 1) cdec[c * H_ + h] = expf(sbuf[i]);
}

// Bb/Cb in chunked layout [c][g][i][n] (bf16, N contiguous)
__global__ void cvt_bc_kernel(const float* __restrict__ xBCc,
                              __bf16* __restrict__ Bb, __bf16* __restrict__ Cb) {
    long long idx = (long long)blockIdx.x * blockDim.x + threadIdx.x;
    if (idx >= (long long)L_ * G_ * N_) return;
    int n = (int)(idx & (N_ - 1));
    int i = (int)((idx >> 7) & (CS_ - 1));
    int g = (int)((idx >> 15) & (G_ - 1));
    int c = (int)(idx >> 18);
    int l = c * CS_ + i;
    Bb[idx] = f2bf(xBCc[(size_t)l * CONVC + INTER + g * N_ + n]);
    Cb[idx] = f2bf(xBCc[(size_t)l * CONVC + INTER + G_ * N_ + g * N_ + n]);
}

// xT bf16 layout [c][h][p][i]  (K = sequence contiguous, for the SSD GEMMs)
__global__ void xT_kernel(const float* __restrict__ xBCc, __bf16* __restrict__ xT) {
    long long idx = (long long)blockIdx.x * blockDim.x + threadIdx.x;
    if (idx >= (long long)NC_ * H_ * P_ * CS_) return;
    int i = (int)(idx & (CS_ - 1));
    int p = (int)((idx >> 8) & (P_ - 1));
    int h = (int)((idx >> 14) & (H_ - 1));
    int c = (int)(idx >> 21);
    int l = c * CS_ + i;
    xT[idx] = f2bf(xBCc[(size_t)l * CONVC + h * P_ + p]);
}

// apply decay mask + dt to raw C.B^T scores; bf16 output [c][h][i][j]
__global__ void mask_scores_kernel(const float* __restrict__ Sg,
                                   const float* __restrict__ cum,
                                   const float* __restrict__ dt_sp,
                                   __bf16* __restrict__ Sm) {
    long long idx = (long long)blockIdx.x * blockDim.x + threadIdx.x;
    if (idx >= (long long)NC_ * H_ * CS_ * CS_) return;
    int j = (int)(idx & (CS_ - 1));
    int i = (int)((idx >> 8) & (CS_ - 1));
    int h = (int)((idx >> 16) & (H_ - 1));
    int c = (int)(idx >> 23);
    float s = 0.f;
    if (j <= i) {
        int g  = h >> 4;                            // rep = H/G = 16
        int il = c * CS_ + i, jl = c * CS_ + j;
        float raw = Sg[(((size_t)(c * G_ + g)) * CS_ + i) * CS_ + j];
        s = raw * expf(cum[(size_t)il * H_ + h] - cum[(size_t)jl * H_ + h])
                * dt_sp[(size_t)jl * H_ + h];
    }
    Sm[idx] = f2bf(s);
}

// decay/dt-weighted B, transposed: wBT bf16 [c][h][n][j]
__global__ void wBT_kernel(const float* __restrict__ xBCc,
                           const float* __restrict__ cum,
                           const float* __restrict__ dt_sp,
                           __bf16* __restrict__ wBT) {
    long long idx = (long long)blockIdx.x * blockDim.x + threadIdx.x;
    if (idx >= (long long)NC_ * H_ * N_ * CS_) return;
    int j = (int)(idx & (CS_ - 1));
    int n = (int)((idx >> 8) & (N_ - 1));
    int h = (int)((idx >> 15) & (H_ - 1));
    int c = (int)(idx >> 22);
    int g  = h >> 4;
    int jl = c * CS_ + j;
    float cl = cum[((size_t)(c * CS_ + CS_ - 1)) * H_ + h];
    float w  = dt_sp[(size_t)jl * H_ + h] * expf(cl - cum[(size_t)jl * H_ + h]);
    wBT[idx] = f2bf(w * xBCc[(size_t)jl * CONVC + INTER + g * N_ + n]);
}

// C * exp(cum) (bf16) [c][h][i][n]
__global__ void cexp_kernel(const float* __restrict__ xBCc,
                            const float* __restrict__ cum,
                            __bf16* __restrict__ Cexp) {
    long long idx = (long long)blockIdx.x * blockDim.x + threadIdx.x;
    if (idx >= (long long)NC_ * H_ * CS_ * N_) return;
    int n = (int)(idx & (N_ - 1));
    int i = (int)((idx >> 7) & (CS_ - 1));
    int h = (int)((idx >> 15) & (H_ - 1));
    int c = (int)(idx >> 22);
    int g  = h >> 4;
    int il = c * CS_ + i;
    Cexp[idx] = f2bf(xBCc[(size_t)il * CONVC + INTER + G_ * N_ + g * N_ + n]
                     * expf(cum[(size_t)il * H_ + h]));
}

// inter-chunk recurrence: prev[c] = cdec[c-1]*prev[c-1] + states[c-1]
__global__ void scan_kernel(const float* __restrict__ states,
                            const float* __restrict__ cdec,
                            __bf16* __restrict__ prevb) {
    int h = blockIdx.x;
    int t = threadIdx.x;                 // 256 threads, 32 elems each, coalesced
    float p[32];
#pragma unroll
    for (int e = 0; e < 32; ++e) p[e] = 0.f;
    for (int c = 0; c < NC_; ++c) {
        size_t base = ((size_t)c * H_ + h) * (P_ * N_);
#pragma unroll
        for (int e = 0; e < 32; ++e)
            prevb[base + e * 256 + t] = f2bf(p[e]);
        float d = cdec[c * H_ + h];
#pragma unroll
        for (int e = 0; e < 32; ++e)
            p[e] = d * p[e] + states[base + e * 256 + t];
    }
}

// y += D*x, gate with silu, grouped RMSNorm over GS=1024, emit bf16 for out_proj
__global__ void gate_norm_kernel(const float* __restrict__ y,     // [c][h][i][p]
                                 const float* __restrict__ proj,
                                 const float* __restrict__ xBCc,
                                 const float* __restrict__ Dp,
                                 const float* __restrict__ norm_w,
                                 __bf16* __restrict__ Gnb) {
    int l = blockIdx.x / G_;
    int g = blockIdx.x % G_;
    int t = threadIdx.x;                 // 256
    int c = l / CS_, i = l % CS_;
    float vals[4];
    float ss = 0.f;
#pragma unroll
    for (int e = 0; e < 4; ++e) {
        int eg = e * 256 + t;            // 0..1023 inside group
        int ei = g * GS_ + eg;           // 0..INTER-1
        int h  = ei >> 6;                // P = 64
        int p  = ei & 63;
        float yy = y[(((size_t)c * H_ + h) * CS_ + i) * P_ + p]
                 + Dp[h] * xBCc[(size_t)l * CONVC + ei];
        float gate = proj[(size_t)l * PROJ + ei];
        float gg = yy * (gate / (1.f + expf(-gate)));
        vals[e] = gg;
        ss += gg * gg;
    }
    __shared__ float red[256];
    red[t] = ss;
    __syncthreads();
    for (int off = 128; off >= 1; off >>= 1) {
        if (t < off) red[t] += red[t + off];
        __syncthreads();
    }
    float inv = rsqrtf(red[0] / (float)GS_ + EPSF);
#pragma unroll
    for (int e = 0; e < 4; ++e) {
        int eg = e * 256 + t;
        int ei = g * GS_ + eg;
        Gnb[(size_t)l * INTER + ei] = f2bf(vals[e] * inv * norm_w[ei]);
    }
}

// ---------------- host orchestration ----------------
static inline size_t al256(size_t x) { return (x + 255) & ~(size_t)255; }

extern "C" void kernel_launch(void* const* d_in, const int* in_sizes, int n_in,
                              void* d_out, int out_size, void* d_ws, size_t ws_size,
                              hipStream_t stream) {
    const float* input  = (const float*)d_in[0];
    const float* w_in   = (const float*)d_in[1];
    const float* conv_w = (const float*)d_in[2];
    const float* conv_b = (const float*)d_in[3];
    const float* dtb    = (const float*)d_in[4];
    const float* A_log  = (const float*)d_in[5];
    const float* Dp     = (const float*)d_in[6];
    const float* norm_w = (const float*)d_in[7];
    const float* w_out  = (const float*)d_in[8];
    float* out = (float*)d_out;

    char* ws = (char*)d_ws;
    size_t off = 0;
    auto take = [&](size_t bytes) { char* p = ws + off; off = al256(off + bytes); return p; };

    __bf16* Xb    = (__bf16*)take((size_t)L_ * HIDN * 2);
    __bf16* W1b   = (__bf16*)take((size_t)PROJ * HIDN * 2);
    __bf16* W2b   = (__bf16*)take((size_t)HIDN * INTER * 2);
    float*  proj  = (float*) take((size_t)L_ * PROJ * 4);
    float*  xBCc  = (float*) take((size_t)L_ * CONVC * 4);
    float*  dt_sp = (float*) take((size_t)L_ * H_ * 4);
    float*  cum   = (float*) take((size_t)L_ * H_ * 4);
    float*  cdec  = (float*) take((size_t)NC_ * H_ * 4);
    __bf16* Bb    = (__bf16*)take((size_t)L_ * G_ * N_ * 2);
    __bf16* Cb    = (__bf16*)take((size_t)L_ * G_ * N_ * 2);
    __bf16* xT    = (__bf16*)take((size_t)NC_ * H_ * P_ * CS_ * 2);
    float*  Sg    = (float*) take((size_t)NC_ * G_ * CS_ * CS_ * 4);
    __bf16* Sm    = (__bf16*)take((size_t)NC_ * H_ * CS_ * CS_ * 2);
    __bf16* wBT   = (__bf16*)take((size_t)NC_ * H_ * N_ * CS_ * 2);
    float*  states= (float*) take((size_t)NC_ * H_ * P_ * N_ * 4);
    __bf16* prevb = (__bf16*)take((size_t)NC_ * H_ * P_ * N_ * 2);
    __bf16* Cexp  = (__bf16*)take((size_t)NC_ * H_ * CS_ * N_ * 2);
    float*  y     = (float*) take((size_t)NC_ * H_ * CS_ * P_ * 4);
    __bf16* Gnb   = (__bf16*)take((size_t)L_ * INTER * 2);

    auto gemm = [&](const __bf16* A, const __bf16* B, float* C,
                    int M, int N, int Kd, int lda, int ldb, int ldc,
                    long long sA, long long sB, long long sC, int batch, int accum) {
        dim3 grid((M + BM - 1) / BM, (N + BN - 1) / BN, batch);
        hipLaunchKernelGGL(gemm_bf16_nt, grid, dim3(256), 0, stream,
                           A, B, C, M, N, Kd, lda, ldb, ldc, sA, sB, sC, accum);
    };
    auto ew = [&](long long n) { return (unsigned)((n + 255) / 256); };

    // 1) precision downconversion for the matrix pipes
    hipLaunchKernelGGL(cvt_f32_bf16, dim3(2048), dim3(256), 0, stream, input, Xb,  (long long)L_ * HIDN);
    hipLaunchKernelGGL(cvt_f32_bf16, dim3(4096), dim3(256), 0, stream, w_in,  W1b, (long long)PROJ * HIDN);
    hipLaunchKernelGGL(cvt_f32_bf16, dim3(4096), dim3(256), 0, stream, w_out, W2b, (long long)HIDN * INTER);

    // 2) in_proj: [2048 x 18560] = X[2048 x 4096] @ W1^T
    gemm(Xb, W1b, proj, L_, PROJ, HIDN, HIDN, HIDN, PROJ, 0, 0, 0, 1, 0);

    // 3) causal depthwise conv + silu
    hipLaunchKernelGGL(conv_silu_kernel, dim3(ew((long long)L_ * CONVC)), dim3(256), 0, stream,
                       proj, conv_w, conv_b, xBCc);

    // 4) dt softplus + chunk cumsum
    hipLaunchKernelGGL(dt_cum_kernel, dim3(NC_ * H_), dim3(CS_), 0, stream,
                       proj, dtb, A_log, dt_sp, cum, cdec);

    // 5) operand repacks (K-contiguous bf16)
    hipLaunchKernelGGL(cvt_bc_kernel, dim3(ew((long long)L_ * G_ * N_)), dim3(256), 0, stream, xBCc, Bb, Cb);
    hipLaunchKernelGGL(xT_kernel, dim3(ew((long long)NC_ * H_ * P_ * CS_)), dim3(256), 0, stream, xBCc, xT);

    // 6) scores: per (chunk, group)  Sg[256x256] = C @ B^T   (K = N = 128)
    gemm(Cb, Bb, Sg, CS_, CS_, N_, N_, N_, CS_,
         (long long)CS_ * N_, (long long)CS_ * N_, (long long)CS_ * CS_, NC_ * G_, 0);

    // 7) decay mask + dt weighting -> bf16 scores per (chunk, head)
    hipLaunchKernelGGL(mask_scores_kernel, dim3(ew((long long)NC_ * H_ * CS_ * CS_)), dim3(256), 0, stream,
                       Sg, cum, dt_sp, Sm);

    // 8) weighted-B transpose for state build
    hipLaunchKernelGGL(wBT_kernel, dim3(ew((long long)NC_ * H_ * N_ * CS_)), dim3(256), 0, stream,
                       xBCc, cum, dt_sp, wBT);

    // 9) states: per (c,h)  [64 x 128] = xT[64 x 256] @ wBT^T  (K = CS = 256)
    gemm(xT, wBT, states, P_, N_, CS_, CS_, CS_, N_,
         (long long)P_ * CS_, (long long)N_ * CS_, (long long)P_ * N_, NC_ * H_, 0);

    // 10) inter-chunk recurrence (8 steps) -> prev states (bf16)
    hipLaunchKernelGGL(scan_kernel, dim3(H_), dim3(256), 0, stream, states, cdec, prevb);

    // 11) C*exp(cum) operand
    hipLaunchKernelGGL(cexp_kernel, dim3(ew((long long)NC_ * H_ * CS_ * N_)), dim3(256), 0, stream,
                       xBCc, cum, Cexp);

    // 12) y_intra: per (c,h)  [256 x 64] = Sm[256 x 256] @ xT^T  (K = 256)
    gemm(Sm, xT, y, CS_, P_, CS_, CS_, CS_, P_,
         (long long)CS_ * CS_, (long long)P_ * CS_, (long long)CS_ * P_, NC_ * H_, 0);

    // 13) y_inter (accumulate): per (c,h)  [256 x 64] += Cexp[256 x 128] @ prev^T (K = 128)
    gemm(Cexp, prevb, y, CS_, P_, N_, N_, N_, P_,
         (long long)CS_ * N_, (long long)P_ * N_, (long long)CS_ * P_, NC_ * H_, 1);

    // 14) +D*x, silu gate, grouped RMSNorm -> bf16 activations
    hipLaunchKernelGGL(gate_norm_kernel, dim3(L_ * G_), dim3(256), 0, stream,
                       y, proj, xBCc, Dp, norm_w, Gnb);

    // 15) out_proj: [2048 x 4096] = Gn[2048 x 8192] @ W2^T
    gemm(Gnb, W2b, out, L_, HIDN, INTER, INTER, INTER, HIDN, 0, 0, 0, 1, 0);

    (void)in_sizes; (void)n_in; (void)out_size; (void)ws_size;
}
